// SlidingWindowCausalMHA_77369540870174
// MI455X (gfx1250) — matrix operand order repaired and checked by reading, not compiled, not run on hardware
//
#include <hip/hip_runtime.h>
#include <stdint.h>

#define SEQ_L   2048
#define DMODEL  1024
#define NHEADS  16
#define DH      64
#define WINSZ   64

typedef __attribute__((ext_vector_type(16))) __bf16   v16bf;
typedef __attribute__((ext_vector_type(8)))  float    v8f;
typedef __attribute__((ext_vector_type(8)))  uint16_t v8u16;
typedef __attribute__((ext_vector_type(4)))  int      v4i;

#if __has_builtin(__builtin_amdgcn_global_load_async_to_lds_b128) && \
    __has_builtin(__builtin_amdgcn_s_wait_asynccnt)
#define HAVE_ASYNC 1
#else
#define HAVE_ASYNC 0
#endif

#if HAVE_ASYNC
typedef __attribute__((address_space(1))) v4i* gv4_ptr;   // global 16B payload
typedef __attribute__((address_space(3))) v4i* lv4_ptr;   // LDS 16B payload
// per-lane 16B async copy: LDS[l] = MEM[g]; tracked by ASYNCcnt
__device__ __forceinline__ void async_cp16(const void* g, void* l) {
    __builtin_amdgcn_global_load_async_to_lds_b128((gv4_ptr)g, (lv4_ptr)l, 0, 0);
}
#endif

// ---- f32 -> bf16 (round-to-nearest-even), pure integer path ----
__device__ __forceinline__ uint16_t f2bf(float f) {
    union { float f; uint32_t u; } x; x.f = f;
    uint32_t r = x.u + 0x7FFFu + ((x.u >> 16) & 1u);
    return (uint16_t)(r >> 16);
}

// Load a 16xbf16 per-lane WMMA fragment from two 16-byte chunks.
__device__ __forceinline__ v16bf ld_frag(const uint16_t* p0, const uint16_t* p1) {
    union { v16bf f; struct { v8u16 lo, hi; } h; } u;
    u.h.lo = *(const v8u16*)p0;
    u.h.hi = *(const v8u16*)p1;
    return u.f;
}

__device__ __forceinline__ v8f vzero8() {
    v8f z = {0.f, 0.f, 0.f, 0.f, 0.f, 0.f, 0.f, 0.f};
    return z;
}

// =============== f32 -> bf16 conversion (4 elems/thread) ===============
__global__ __launch_bounds__(256)
void cvt_f32_bf16(const float* __restrict__ in, uint16_t* __restrict__ out, int n4) {
    int i = blockIdx.x * blockDim.x + threadIdx.x;
    if (i >= n4) return;
    float4 v = ((const float4*)in)[i];
    ushort4 r;
    r.x = f2bf(v.x); r.y = f2bf(v.y); r.z = f2bf(v.z); r.w = f2bf(v.w);
    ((ushort4*)out)[i] = r;
}

// ====== f32 -> bf16 with transpose: out[c*R + r] = bf16(in[r*C + c]) ======
// 32x32 tile through LDS; 256 threads as 32x8.
__global__ __launch_bounds__(256)
void cvt_transpose_f32_bf16(const float* __restrict__ in, uint16_t* __restrict__ out,
                            int R, int C) {
    __shared__ uint16_t t[32][33];
    const int r0 = blockIdx.y * 32, c0 = blockIdx.x * 32;
    const int tx = threadIdx.x & 31, ty = threadIdx.x >> 5;
#pragma unroll
    for (int i = 0; i < 4; ++i) {
        int rr = ty + i * 8;
        t[rr][tx] = f2bf(in[(size_t)(r0 + rr) * C + c0 + tx]);
    }
    __syncthreads();
#pragma unroll
    for (int i = 0; i < 4; ++i) {
        int rr = ty + i * 8;
        out[(size_t)(c0 + rr) * R + r0 + tx] = t[tx][rr];
    }
}

// =============== bf16 GEMM: C(MxN) = A(MxK) * Bt^T, Bt is NxK row-major =======
// Block tile 128x128, 8 waves of wave32, each wave -> 32x64 (2x4 WMMA frags).
// Double-buffered LDS staging via GLOBAL_LOAD_ASYNC_TO_LDS when available.
template<bool OUT_BF16>
__global__ __launch_bounds__(256)
void gemm_bf16_wmma(const uint16_t* __restrict__ A,
                    const uint16_t* __restrict__ Bt,
                    void* __restrict__ Cv,
                    int M, int N, int K) {
    __shared__ __attribute__((aligned(16))) uint16_t As[2][128 * 40];  // [m][k] pad 40
    __shared__ __attribute__((aligned(16))) uint16_t Bs[2][128 * 40];  // [n][k] pad 40

    const int tid  = threadIdx.x;
    const int lane = tid & 31;
    const int w    = tid >> 5;         // 0..7
    const int wm   = (w & 3) * 32;     // wave M offset in block tile
    const int wn   = (w >> 2) * 64;    // wave N offset in block tile
    const int bm   = blockIdx.y * 128;
    const int bn   = blockIdx.x * 128;
    const int hl   = lane >> 4;        // half-wave select
    const int ll   = lane & 15;

    v8f acc[2][4];
#pragma unroll
    for (int mi = 0; mi < 2; ++mi)
#pragma unroll
        for (int ni = 0; ni < 4; ++ni) acc[mi][ni] = vzero8();

    // Stage one 128x32 slab of A and of Bt into LDS buffer `buf`.
    auto stage = [&](int buf, int k0) {
#pragma unroll
        for (int i = 0; i < 2; ++i) {
            int ch = tid + i * 256;          // 512 chunks of 8 elems per matrix
            int r  = ch >> 2, co = (ch & 3) * 8;
            const uint16_t* ga = A  + (size_t)(bm + r) * K + k0 + co;
            const uint16_t* gb = Bt + (size_t)(bn + r) * K + k0 + co;
#if HAVE_ASYNC
            async_cp16(ga, &As[buf][r * 40 + co]);
            async_cp16(gb, &Bs[buf][r * 40 + co]);
#else
            *(v8u16*)&As[buf][r * 40 + co] = *(const v8u16*)ga;
            *(v8u16*)&Bs[buf][r * 40 + co] = *(const v8u16*)gb;
#endif
        }
    };

    const int nIter = K / 32;
    stage(0, 0);
    for (int it = 0; it < nIter; ++it) {
        const int buf = it & 1;
#if HAVE_ASYNC
        __builtin_amdgcn_s_wait_asynccnt(0);
#endif
        __syncthreads();                       // current buffer published
        if (it + 1 < nIter) stage(buf ^ 1, (it + 1) * 32);  // overlap next slab
        if (it + 2 < nIter) {                  // L2 prefetch two slabs ahead
            __builtin_prefetch(A  + (size_t)(bm + (tid >> 1)) * K + (it + 2) * 32, 0, 1);
            __builtin_prefetch(Bt + (size_t)(bn + (tid >> 1)) * K + (it + 2) * 32, 0, 1);
        }

        // ---- compute: 8 WMMAs per wave per K-slab ----
        v16bf af[2];
#pragma unroll
        for (int mi = 0; mi < 2; ++mi) {
            const uint16_t* ap = &As[buf][(wm + mi * 16 + ll) * 40 + hl * 8];
            af[mi] = ld_frag(ap, ap + 16);   // K {0..7,16..23} / {8..15,24..31}
        }
#pragma unroll
        for (int ni = 0; ni < 4; ++ni) {
            const uint16_t* bp = &Bs[buf][(wn + ni * 16 + ll) * 40 + hl * 8];
            v16bf bfr = ld_frag(bp, bp + 16); // K {0..7,16..23} / {8..15,24..31}
#pragma unroll
            for (int mi = 0; mi < 2; ++mi) {
                acc[mi][ni] = __builtin_amdgcn_wmma_f32_16x16x32_bf16(
                    false, af[mi], false, bfr, (short)0, acc[mi][ni], false, false);
            }
        }
        // no trailing barrier: next iteration's wait+barrier protects buffer reuse
    }

    // ---- store C (D layout: row = hl*8+g, col = ll) ----
#pragma unroll
    for (int mi = 0; mi < 2; ++mi) {
#pragma unroll
        for (int ni = 0; ni < 4; ++ni) {
            int row0 = bm + wm + mi * 16 + hl * 8;
            int col  = bn + wn + ni * 16 + ll;
#pragma unroll
            for (int g = 0; g < 8; ++g) {
                if (OUT_BF16)
                    ((uint16_t*)Cv)[(size_t)(row0 + g) * N + col] = f2bf(acc[mi][ni][g]);
                else
                    ((float*)Cv)[(size_t)(row0 + g) * N + col] = acc[mi][ni][g];
            }
        }
    }
}

// =============== sliding-window attention, WMMA flash-style ===============
// grid: (L/64, H). block: 128 threads = 4 waves; wave w owns 16 queries.
// Block key window: [q0-80, q0+63] -> 144 keys, per-wave 96-key padded window.
__global__ __launch_bounds__(128)
void attn_wmma(const uint16_t* __restrict__ Q,
               const uint16_t* __restrict__ Kb,
               const uint16_t* __restrict__ Vb,
               uint16_t* __restrict__ Ctx) {
    __shared__ __attribute__((aligned(16))) uint16_t Ks[144 * 72];      // [key][d] pad 72
    __shared__ __attribute__((aligned(16))) uint16_t Vt[64 * 152];      // [d][key] pad 152
    __shared__ __attribute__((aligned(16))) uint16_t Ps[4][16 * 104];   // per-wave P, pad 104

    const int h     = blockIdx.y;
    const int q0    = blockIdx.x * 64;
    const int tid   = threadIdx.x;
    const int lane  = tid & 31;
    const int w     = tid >> 5;       // 0..3
    const int hl    = lane >> 4;
    const int ll    = lane & 15;
    const int kbase = q0 - 80;        // absolute key index of relative key 0

    // ---- stage K (async direct copy) and V (transposed, sync) tiles ----
#pragma unroll
    for (int i = 0; i < 9; ++i) {
        int ch = tid + i * 128;       // 1152 chunks of 8 elems
        int r  = ch >> 3, co = (ch & 7) * 8;
        int krow = kbase + r; if (krow < 0) krow = 0;   // clamped rows get masked later
        const uint16_t* gk = Kb + (size_t)krow * DMODEL + h * DH + co;
#if HAVE_ASYNC
        async_cp16(gk, Ks + r * 72 + co);
#else
        *(v8u16*)(Ks + r * 72 + co) = *(const v8u16*)gk;
#endif
        v8u16 vv = *(const v8u16*)(Vb + (size_t)krow * DMODEL + h * DH + co);
#pragma unroll
        for (int j = 0; j < 8; ++j) Vt[(co + j) * 152 + r] = vv[j];
    }
#if HAVE_ASYNC
    __builtin_amdgcn_s_wait_asynccnt(0);
#endif
    __syncthreads();

    // ---- Q fragments (2 K-chunks of 32 over d_h=64), straight from global ----
    v16bf qf[2];
#pragma unroll
    for (int c = 0; c < 2; ++c) {
        const uint16_t* qp =
            Q + (size_t)(q0 + w * 16 + ll) * DMODEL + h * DH + c * 32 + hl * 8;
        qf[c] = ld_frag(qp, qp + 16);
    }

    // ---- scores: 6 key tiles x 2 WMMAs ----
    v8f s[6];
#pragma unroll
    for (int t = 0; t < 6; ++t) {
        int kt = w + t;               // relative key tile in block window
        v8f a = vzero8();
#pragma unroll
        for (int c = 0; c < 2; ++c) {
            const uint16_t* kp = Ks + (kt * 16 + ll) * 72 + c * 32 + hl * 8;
            v16bf kf = ld_frag(kp, kp + 16);
            a = __builtin_amdgcn_wmma_f32_16x16x32_bf16(
                false, qf[c], false, kf, (short)0, a, false, false);
        }
        s[t] = a;
    }

    // ---- mask + softmax (row lives across 16 lanes of one half-wave) ----
    const float scale = 0.125f;       // 64^-0.5
    const float NEG = -__builtin_inff();
#pragma unroll
    for (int g = 0; g < 8; ++g) {
        int qi = q0 + w * 16 + hl * 8 + g;
        float rmax = NEG;
#pragma unroll
        for (int t = 0; t < 6; ++t) {
            int kj = kbase + (w + t) * 16 + ll;
            float v = s[t][g] * scale;
            bool ok = (kj >= 0) & (kj <= qi) & (kj >= qi - (WINSZ - 1));
            v = ok ? v : NEG;
            s[t][g] = v;
            rmax = fmaxf(rmax, v);
        }
        rmax = fmaxf(rmax, __shfl_xor(rmax, 1, 32));
        rmax = fmaxf(rmax, __shfl_xor(rmax, 2, 32));
        rmax = fmaxf(rmax, __shfl_xor(rmax, 4, 32));
        rmax = fmaxf(rmax, __shfl_xor(rmax, 8, 32));
        float rsum = 0.f;
#pragma unroll
        for (int t = 0; t < 6; ++t) {
            float p = __expf(s[t][g] - rmax);
            s[t][g] = p;
            rsum += p;
        }
        rsum += __shfl_xor(rsum, 1, 32);
        rsum += __shfl_xor(rsum, 2, 32);
        rsum += __shfl_xor(rsum, 4, 32);
        rsum += __shfl_xor(rsum, 8, 32);
        float inv = 1.0f / rsum;      // diagonal key always valid -> rsum > 0
#pragma unroll
        for (int t = 0; t < 6; ++t)
            Ps[w][(hl * 8 + g) * 104 + t * 16 + ll] = f2bf(s[t][g] * inv);
    }
    // per-wave LDS region; same-wave DS ops are in order -> no barrier needed

    // ---- P fragments (A layout, 3 chunks over 96 keys) ----
    v16bf pf[3];
#pragma unroll
    for (int c = 0; c < 3; ++c) {
        const uint16_t* pp = &Ps[w][ll * 104 + c * 32 + hl * 8];
        pf[c] = ld_frag(pp, pp + 16);
    }

    // ---- ctx = P @ V : 4 d-tiles x 3 WMMAs, B from transposed V tile ----
#pragma unroll
    for (int nt = 0; nt < 4; ++nt) {
        v8f a = vzero8();
#pragma unroll
        for (int c = 0; c < 3; ++c) {
            const uint16_t* vp = Vt + (nt * 16 + ll) * 152 + w * 16 + c * 32 + hl * 8;
            v16bf vf = ld_frag(vp, vp + 16);
            a = __builtin_amdgcn_wmma_f32_16x16x32_bf16(
                false, pf[c], false, vf, (short)0, a, false, false);
        }
#pragma unroll
        for (int g = 0; g < 8; ++g) {
            int qi = q0 + w * 16 + hl * 8 + g;
            Ctx[(size_t)qi * DMODEL + h * DH + nt * 16 + ll] = f2bf(a[g]);
        }
    }
}

// =============== host-side launch ===============
extern "C" void kernel_launch(void* const* d_in, const int* in_sizes, int n_in,
                              void* d_out, int out_size, void* d_ws, size_t ws_size,
                              hipStream_t stream) {
    (void)in_sizes; (void)n_in; (void)out_size; (void)ws_size;
    const float* x  = (const float*)d_in[0];
    const float* Wq = (const float*)d_in[1];
    const float* Wk = (const float*)d_in[2];
    const float* Wv = (const float*)d_in[3];
    const float* Wo = (const float*)d_in[4];
    float* out = (float*)d_out;

    uint8_t* ws = (uint8_t*)d_ws;
    size_t off = 0;
    auto carve = [&](size_t bytes) -> uint16_t* {
        uint16_t* p = (uint16_t*)(ws + off);
        off += (bytes + 255) & ~(size_t)255;
        return p;
    };
    const size_t actB = (size_t)SEQ_L * DMODEL * 2;
    const size_t wB   = (size_t)DMODEL * DMODEL * 2;
    uint16_t* xb   = carve(actB);
    uint16_t* wqt  = carve(wB);   // transposed bf16 weights (NxK)
    uint16_t* wkt  = carve(wB);
    uint16_t* wvt  = carve(wB);
    uint16_t* wot  = carve(wB);
    uint16_t* qb   = carve(actB);
    uint16_t* kb   = carve(actB);
    uint16_t* vb   = carve(actB);
    uint16_t* ctxb = carve(actB);

    // bf16 conversions (x plain; weights transposed for async-friendly staging)
    {
        int n4a = SEQ_L * DMODEL / 4;
        cvt_f32_bf16<<<(n4a + 255) / 256, 256, 0, stream>>>(x, xb, n4a);
        dim3 tg(DMODEL / 32, DMODEL / 32);   // (32,32)
        cvt_transpose_f32_bf16<<<tg, 256, 0, stream>>>(Wq, wqt, DMODEL, DMODEL);
        cvt_transpose_f32_bf16<<<tg, 256, 0, stream>>>(Wk, wkt, DMODEL, DMODEL);
        cvt_transpose_f32_bf16<<<tg, 256, 0, stream>>>(Wv, wvt, DMODEL, DMODEL);
        cvt_transpose_f32_bf16<<<tg, 256, 0, stream>>>(Wo, wot, DMODEL, DMODEL);
    }

    // Q/K/V projections (bf16 out)
    dim3 gg(DMODEL / 128, SEQ_L / 128);   // (8,16)
    gemm_bf16_wmma<true><<<gg, 256, 0, stream>>>(xb, wqt, qb, SEQ_L, DMODEL, DMODEL);
    gemm_bf16_wmma<true><<<gg, 256, 0, stream>>>(xb, wkt, kb, SEQ_L, DMODEL, DMODEL);
    gemm_bf16_wmma<true><<<gg, 256, 0, stream>>>(xb, wvt, vb, SEQ_L, DMODEL, DMODEL);

    // windowed attention
    attn_wmma<<<dim3(SEQ_L / 64, NHEADS), 128, 0, stream>>>(qb, kb, vb, ctxb);

    // output projection (f32 out)
    gemm_bf16_wmma<false><<<gg, 256, 0, stream>>>(ctxb, wot, out, SEQ_L, DMODEL, DMODEL);
}
